// SimpleGraphSAGE_88768384074310
// MI455X (gfx1250) — matrix operand order repaired and checked by reading, not compiled, not run on hardware
//
#include <hip/hip_runtime.h>

// ---------------------------------------------------------------------------
// GraphSAGE forward for MI455X (gfx1250): wave32, bf16 WMMA GEMMs with
// double-buffered GLOBAL_LOAD_ASYNC_TO_LDS staging (ASYNCcnt pipeline),
// fp32 atomic scatter-aggregation for the irregular graph part.
// ---------------------------------------------------------------------------

typedef __attribute__((ext_vector_type(16))) __bf16 v16bf;
typedef __attribute__((ext_vector_type(8)))  __bf16 v8bf;
typedef __attribute__((ext_vector_type(8)))  float  v8f;

#define N_NODES 100000
#define NUM_EDGES 3200000
#define IN_DIM 128
#define HID 1024
#define N_CLS 153
#define MPAD 100096   // 782 * 128, row padding for 128-row GEMM tiles

#if __has_builtin(__builtin_amdgcn_s_wait_asynccnt)
#define WAIT_ASYNC(n) __builtin_amdgcn_s_wait_asynccnt(n)
#else
#define WAIT_ASYNC(n) asm volatile("s_wait_asynccnt %0" ::"i"(n) : "memory")
#endif

// Async DMA: 16 bytes per lane, global -> LDS, tracked by ASYNCcnt.
// VDST = per-lane LDS byte offset (wave-relative), VADDR = 64-bit address.
__device__ __forceinline__ void async_ld_b128(unsigned lds_off,
                                              const void* gptr) {
    asm volatile("global_load_async_to_lds_b128 %0, %1, off"
                 :
                 : "v"(lds_off), "v"((unsigned long long)(uintptr_t)gptr)
                 : "memory");
}

__device__ __forceinline__ unsigned lds_off32(const void* p) {
    // generic pointer to LDS = {aperture_hi32, lds_byte_offset32}
    return (unsigned)(uintptr_t)p;
}

// ---------------------------------------------------------------------------
// Degree count: one thread per edge, fp32 atomic.
// ---------------------------------------------------------------------------
__global__ void k_count_deg(const long long* __restrict__ dst,
                            float* __restrict__ deg, int nE) {
    int e = blockIdx.x * blockDim.x + threadIdx.x;
    if (e < nE) atomicAdd(&deg[dst[e]], 1.0f);
}

__global__ void k_recip(const float* __restrict__ deg,
                        float* __restrict__ dinv, int n) {
    int i = blockIdx.x * blockDim.x + threadIdx.x;
    if (i < n) dinv[i] = 1.0f / fmaxf(deg[i], 1.0f);
}

// ---------------------------------------------------------------------------
// Edge aggregation: one wave (32 lanes) per edge; each lane moves a float4
// (32 lanes * 4 = 128 features per chunk), atomic-adds into msg[dst].
// ---------------------------------------------------------------------------
__global__ __launch_bounds__(256) void k_aggregate(
    const float* __restrict__ feat,
    const long long* __restrict__ src,
    const long long* __restrict__ dst,
    float* __restrict__ msg, int nE, int D) {
    int e = blockIdx.x * 8 + (threadIdx.x >> 5);
    if (e >= nE) return;
    int lane = threadIdx.x & 31;
    long long s = src[e], d = dst[e];
    const float4* fs = (const float4*)(feat + (size_t)s * D) + lane;
    float* md = msg + (size_t)d * D + lane * 4;
    int chunks = D >> 7;  // D / 128
    for (int c = 0; c < chunks; ++c) {
        float4 v = fs[c * 32];
        float* p = md + c * 128;
        atomicAdd(p + 0, v.x);
        atomicAdd(p + 1, v.y);
        atomicAdd(p + 2, v.z);
        atomicAdd(p + 3, v.w);
    }
}

// ---------------------------------------------------------------------------
// Weight prep: Wout[r, 0:Din] = Wl, Wout[r, Din:2*Din] = Wr   (fp32 -> bf16)
// ---------------------------------------------------------------------------
__global__ void k_prep_w_concat(const float* __restrict__ Wl,
                                const float* __restrict__ Wr,
                                __bf16* __restrict__ Wout, int rows, int Din) {
    long long i = (long long)blockIdx.x * blockDim.x + threadIdx.x;
    long long total = (long long)rows * (2 * Din);
    if (i >= total) return;
    int ld = 2 * Din;
    int r = (int)(i / ld), c = (int)(i % ld);
    float v = (c < Din) ? Wl[(size_t)r * Din + c]
                        : Wr[(size_t)r * Din + (c - Din)];
    Wout[i] = (__bf16)v;
}

// Classifier weight: convert + zero-pad rows to rowsPad.
__global__ void k_prep_w_pad(const float* __restrict__ Wsrc,
                             __bf16* __restrict__ Wout,
                             int rowsReal, int rowsPad, int K) {
    long long i = (long long)blockIdx.x * blockDim.x + threadIdx.x;
    long long total = (long long)rowsPad * K;
    if (i >= total) return;
    int r = (int)(i / K);
    Wout[i] = (r < rowsReal) ? (__bf16)Wsrc[i] : (__bf16)0.0f;
}

// Layer-1 A-matrix: A[m, 0:Din] = msg*dinv, A[m, Din:2Din] = x   (bf16)
__global__ void k_prep_A(const float* __restrict__ msg,
                         const float* __restrict__ dinv,
                         const float* __restrict__ x,
                         __bf16* __restrict__ Aout, int M, int Din) {
    long long i = (long long)blockIdx.x * blockDim.x + threadIdx.x;
    long long total = (long long)M * (2 * Din);
    if (i >= total) return;
    int ld = 2 * Din;
    int m = (int)(i / ld), c = (int)(i % ld);
    float v = (c < Din) ? msg[(size_t)m * Din + c] * dinv[m]
                        : x[(size_t)m * Din + (c - Din)];
    Aout[i] = (__bf16)v;
}

// Layer-2: fill only the aggregated half (x-half is written by GEMM1 epilogue)
__global__ void k_prep_msg(const float* __restrict__ msg,
                           const float* __restrict__ dinv,
                           __bf16* __restrict__ Aout, int M, int Din, int ldA) {
    long long i = (long long)blockIdx.x * blockDim.x + threadIdx.x;
    long long total = (long long)M * Din;
    if (i >= total) return;
    int m = (int)(i / Din), c = (int)(i % Din);
    Aout[(size_t)m * ldA + c] = (__bf16)(msg[i] * dinv[m]);
}

// ---------------------------------------------------------------------------
// bf16 WMMA GEMM:  C[M,N] = A[M,K] @ W[N,K]^T + bias
//   Block tile 128x128, 8 waves in 2(rows) x 4(cols); wave tile 64x32
//   = 4x2 tiles of 16x16 f32 accumulators. BK = 32 (one bf16 WMMA K-step).
//   Global->LDS staging via double-buffered GLOBAL_LOAD_ASYNC_TO_LDS_B128.
// MODE 0: ELU, store fp32 to outF (m<Mreal) AND bf16 to outB (+nOffB)
// MODE 1: ELU, store bf16 to outB only
// MODE 2: bias only, store fp32 to outF with m/n bounds
// ---------------------------------------------------------------------------
__device__ __forceinline__ v16bf cat8(v8bf lo, v8bf hi) {
    return __builtin_shufflevector(lo, hi, 0, 1, 2, 3, 4, 5, 6, 7,
                                   8, 9, 10, 11, 12, 13, 14, 15);
}

template <int MODE>
__global__ __launch_bounds__(256) void k_gemm_bf16(
    const __bf16* __restrict__ A, const __bf16* __restrict__ W,
    const float* __restrict__ bias,
    float* __restrict__ outF, __bf16* __restrict__ outB,
    int K, int Mreal, int Nreal, int ldOutF, int ldOutB, int nOffB) {
    __shared__ __attribute__((aligned(64))) __bf16 sA[2][128 * 32];
    __shared__ __attribute__((aligned(64))) __bf16 sB[2][128 * 32];

    const int tid = threadIdx.x;
    const int tileM = blockIdx.y * 128;
    const int tileN = blockIdx.x * 128;
    const int lane = tid & 31;
    const int wv = tid >> 5;
    const int wr = wv & 1;    // wave row: 0..1  (64 rows each)
    const int wc = wv >> 1;   // wave col: 0..3  (32 cols each)

    v8f acc[4][2] = {};

    // staging: thread t owns 16 bf16 (32B = 2 async b128) of row t/2, half t&1
    const int lrow = tid >> 1;
    const int lhalf = (tid & 1) * 16;
    const __bf16* Ag = A + (size_t)(tileM + lrow) * K + lhalf;
    const __bf16* Wg = W + (size_t)(tileN + lrow) * K + lhalf;
    unsigned sAo[2], sBo[2];
    sAo[0] = lds_off32(&sA[0][lrow * 32 + lhalf]);
    sAo[1] = lds_off32(&sA[1][lrow * 32 + lhalf]);
    sBo[0] = lds_off32(&sB[0][lrow * 32 + lhalf]);
    sBo[1] = lds_off32(&sB[1][lrow * 32 + lhalf]);

    // per-lane fragment addressing (wave32; contiguous K per lane half, same for A and B)
    const int r = lane & 15;
    const int kg = (lane >> 4) * 16;   // A: K run {0-15} / {16-31}
    const int kb = (lane >> 4) * 16;   // B: K run {0-15} / {16-31}

    auto issue = [&](int k0, int b) {
        async_ld_b128(sAo[b], Ag + k0);
        async_ld_b128(sAo[b] + 16, Ag + k0 + 8);
        async_ld_b128(sBo[b], Wg + k0);
        async_ld_b128(sBo[b] + 16, Wg + k0 + 8);
    };

    const int ksteps = K >> 5;
    issue(0, 0);
    for (int i = 0; i < ksteps; ++i) {
        if (i + 1 < ksteps) {
            issue((i + 1) << 5, (i + 1) & 1);
            WAIT_ASYNC(4);  // previous batch (4 async ops) has landed
        } else {
            WAIT_ASYNC(0);
        }
        __syncthreads();

        const __bf16* cA = sA[i & 1];
        const __bf16* cB = sB[i & 1];
        v16bf bf0 = *(const v16bf*)&cB[(wc * 32 + 0 + r) * 32 + kb];
        v16bf bf1 = *(const v16bf*)&cB[(wc * 32 + 16 + r) * 32 + kb];
#pragma unroll
        for (int rt = 0; rt < 4; ++rt) {
            const __bf16* ap = &cA[(wr * 64 + rt * 16 + r) * 32 + kg];
            v16bf af = cat8(*(const v8bf*)ap, *(const v8bf*)(ap + 8));
            acc[rt][0] = __builtin_amdgcn_wmma_f32_16x16x32_bf16(
                false, af, false, bf0, (short)0, acc[rt][0], false, false);
            acc[rt][1] = __builtin_amdgcn_wmma_f32_16x16x32_bf16(
                false, af, false, bf1, (short)0, acc[rt][1], false, false);
        }
        __syncthreads();
    }

    // Epilogue. C/D layout: lane = col (0..15 twice); VGPR j -> row j (+8 for
    // upper lane half).
    const int radd = (lane >> 4) << 3;
    const int nloc = lane & 15;
#pragma unroll
    for (int rt = 0; rt < 4; ++rt) {
#pragma unroll
        for (int ct = 0; ct < 2; ++ct) {
            int n = tileN + wc * 32 + ct * 16 + nloc;
            float bv = (n < Nreal) ? bias[n] : 0.0f;
#pragma unroll
            for (int j = 0; j < 8; ++j) {
                int m = tileM + wr * 64 + rt * 16 + radd + j;
                float v = acc[rt][ct][j] + bv;
                if (MODE != 2) v = (v > 0.0f) ? v : (expf(v) - 1.0f);  // ELU
                if (MODE == 0) {
                    if (m < Mreal) outF[(size_t)m * ldOutF + n] = v;
                    outB[(size_t)m * ldOutB + nOffB + n] = (__bf16)v;
                } else if (MODE == 1) {
                    outB[(size_t)m * ldOutB + nOffB + n] = (__bf16)v;
                } else {
                    if (m < Mreal && n < Nreal)
                        outF[(size_t)m * ldOutF + n] = v;
                }
            }
        }
    }
}

// ---------------------------------------------------------------------------
extern "C" void kernel_launch(void* const* d_in, const int* in_sizes, int n_in,
                              void* d_out, int out_size, void* d_ws,
                              size_t ws_size, hipStream_t stream) {
    (void)in_sizes; (void)n_in; (void)out_size; (void)ws_size;

    const float* x        = (const float*)d_in[0];
    const long long* ei   = (const long long*)d_in[1];  // int64 [2, E]
    const long long* src  = ei;
    const long long* dst  = ei + NUM_EDGES;
    const float* W1l = (const float*)d_in[2];
    const float* b1  = (const float*)d_in[3];
    const float* W1r = (const float*)d_in[4];
    const float* W2l = (const float*)d_in[5];
    const float* b2  = (const float*)d_in[6];
    const float* W2r = (const float*)d_in[7];
    const float* Wc  = (const float*)d_in[8];
    const float* bc  = (const float*)d_in[9];

    char* ws = (char*)d_ws;
    size_t off = 0;
    auto take = [&](size_t bytes) {
        char* p = ws + off;
        off += (bytes + 511) & ~(size_t)511;
        return p;
    };
    float*  deg  = (float*)take((size_t)N_NODES * 4);
    float*  dinv = (float*)take((size_t)N_NODES * 4);
    float*  msg1 = (float*)take((size_t)N_NODES * IN_DIM * 4);
    float*  msg2 = (float*)take((size_t)N_NODES * HID * 4);
    float*  h1   = (float*)take((size_t)N_NODES * HID * 4);
    __bf16* A1   = (__bf16*)take((size_t)MPAD * 256 * 2);
    __bf16* A2   = (__bf16*)take((size_t)MPAD * 2048 * 2);
    __bf16* A3   = (__bf16*)take((size_t)MPAD * 1024 * 2);
    __bf16* Wb1  = (__bf16*)take((size_t)HID * 256 * 2);
    __bf16* Wb2  = (__bf16*)take((size_t)HID * 2048 * 2);
    __bf16* Wbc  = (__bf16*)take((size_t)256 * HID * 2);

    // atomically-accumulated buffers must start at zero every call
    hipMemsetAsync(deg, 0, (size_t)N_NODES * 4, stream);
    hipMemsetAsync(msg1, 0, (size_t)N_NODES * IN_DIM * 4, stream);
    hipMemsetAsync(msg2, 0, (size_t)N_NODES * HID * 4, stream);

    // degree + inverse
    k_count_deg<<<(NUM_EDGES + 255) / 256, 256, 0, stream>>>(dst, deg, NUM_EDGES);
    k_recip<<<(N_NODES + 255) / 256, 256, 0, stream>>>(deg, dinv, N_NODES);

    // layer-1 aggregation (128-dim)
    k_aggregate<<<(NUM_EDGES + 7) / 8, 256, 0, stream>>>(x, src, dst, msg1,
                                                         NUM_EDGES, IN_DIM);

    // weight prep (bf16)
    {
        long long t = (long long)HID * 256;
        k_prep_w_concat<<<(int)((t + 255) / 256), 256, 0, stream>>>(
            W1l, W1r, Wb1, HID, IN_DIM);
    }
    {
        long long t = (long long)HID * 2048;
        k_prep_w_concat<<<(int)((t + 255) / 256), 256, 0, stream>>>(
            W2l, W2r, Wb2, HID, HID);
    }
    {
        long long t = 256LL * HID;
        k_prep_w_pad<<<(int)((t + 255) / 256), 256, 0, stream>>>(
            Wc, Wbc, N_CLS, 256, HID);
    }

    // A1 = [msg1*dinv | x] in bf16
    {
        long long t = (long long)N_NODES * 256;
        k_prep_A<<<(int)((t + 255) / 256), 256, 0, stream>>>(msg1, dinv, x, A1,
                                                             N_NODES, IN_DIM);
    }

    // GEMM1: h1 = ELU(A1 @ Wb1^T + b1); also emit bf16 h1 into right half of A2
    {
        dim3 g(HID / 128, MPAD / 128);
        k_gemm_bf16<0><<<g, 256, 0, stream>>>(A1, Wb1, b1, h1, A2,
                                              256, N_NODES, HID, HID, 2048, HID);
    }

    // layer-2 aggregation (1024-dim) over fp32 h1
    k_aggregate<<<(NUM_EDGES + 7) / 8, 256, 0, stream>>>(h1, src, dst, msg2,
                                                         NUM_EDGES, HID);

    // A2 left half = msg2 * dinv (bf16)
    {
        long long t = (long long)N_NODES * HID;
        k_prep_msg<<<(int)((t + 255) / 256), 256, 0, stream>>>(msg2, dinv, A2,
                                                               N_NODES, HID, 2048);
    }

    // GEMM2: A3 = bf16(ELU(A2 @ Wb2^T + b2))
    {
        dim3 g(HID / 128, MPAD / 128);
        k_gemm_bf16<1><<<g, 256, 0, stream>>>(A2, Wb2, b2, nullptr, A3,
                                              2048, N_NODES, HID, 0, 1024, 0);
    }

    // classifier: out = A3 @ Wbc^T + bc  (N padded 153 -> 256, masked store)
    {
        dim3 g(256 / 128, MPAD / 128);
        k_gemm_bf16<2><<<g, 256, 0, stream>>>(A3, Wbc, bc, (float*)d_out,
                                              nullptr, HID, N_NODES, N_CLS,
                                              N_CLS, 0, 0);
    }
}
